// MessageBlock_17815524344046
// MI455X (gfx1250) — matrix operand order repaired and checked by reading, not compiled, not run on hardware
//
#include <hip/hip_runtime.h>

// ---------------- problem constants ----------------
#define Bn   1024      // nodes
#define Cc   128       // channels
#define En   10        // species
#define DIMI 16
#define KT   288       // 256 (x⊗x) + 16 (x) + 1 (bias) + 15 pad  = 9 * 32
#define KS   9         // K-steps of 32 for WMMA

typedef __attribute__((ext_vector_type(16))) __bf16 v16bf;
typedef __attribute__((ext_vector_type(8)))  float  v8f;

// =====================================================================
// Kernel 1: per-species weight-folded contraction tables (bf16)
//   T[e][c][m*16+i1][k] ,  k<256: Σq U3[(m),i1,i2,i,q]·W3[e,q,c]  (k=i*16+i2)
//                          256..271: Σq U2[(m),i1,j,q]·W2[e,q,c]  (j=k-256)
//                          272:      U1[(m),i1]·W1[e,c]
//                          273..287: 0
// =====================================================================
__global__ __launch_bounds__(256)
void build_table(const float* __restrict__ U3_0, const float* __restrict__ U2_0,
                 const float* __restrict__ U1_0, const float* __restrict__ W3_0,
                 const float* __restrict__ W2_0, const float* __restrict__ W1_0,
                 const float* __restrict__ U3_1, const float* __restrict__ U2_1,
                 const float* __restrict__ U1_1, const float* __restrict__ W3_1,
                 const float* __restrict__ W2_1, const float* __restrict__ W1_1,
                 __bf16* __restrict__ tbl)
{
    int ec = blockIdx.x;              // 0 .. E*C-1
    int e = ec / Cc, c = ec % Cc;

    float w3a[5], w3b[7], w2a[2], w2b[3];
    #pragma unroll
    for (int q = 0; q < 5; ++q) w3a[q] = W3_0[(e*5 + q)*Cc + c];
    #pragma unroll
    for (int q = 0; q < 7; ++q) w3b[q] = W3_1[(e*7 + q)*Cc + c];
    #pragma unroll
    for (int q = 0; q < 2; ++q) w2a[q] = W2_0[(e*2 + q)*Cc + c];
    #pragma unroll
    for (int q = 0; q < 3; ++q) w2b[q] = W2_1[(e*3 + q)*Cc + c];
    float w1a = W1_0[e*Cc + c];
    float w1b = W1_1[e*Cc + c];

    __bf16* T = tbl + (size_t)ec * (64 * KT);

    for (int idx = threadIdx.x; idx < 64 * KT; idx += 256) {
        int k  = idx % KT;
        int mi = idx / KT;            // 0..63
        int m  = mi >> 4, i1 = mi & 15;
        float s = 0.0f;
        if (k < 256) {
            int i = k >> 4, i2 = k & 15;
            if (m == 0) {
                const float* u = U3_0 + ((i1*16 + i2)*16 + i)*5;
                #pragma unroll
                for (int q = 0; q < 5; ++q) s += u[q] * w3a[q];
            } else {
                const float* u = U3_1 + ((((m-1)*16 + i1)*16 + i2)*16 + i)*7;
                #pragma unroll
                for (int q = 0; q < 7; ++q) s += u[q] * w3b[q];
            }
        } else if (k < 272) {
            int j = k - 256;
            if (m == 0) {
                const float* u = U2_0 + (i1*16 + j)*2;
                s = u[0]*w2a[0] + u[1]*w2a[1];
            } else {
                const float* u = U2_1 + (((m-1)*16 + i1)*16 + j)*3;
                s = u[0]*w2b[0] + u[1]*w2b[1] + u[2]*w2b[2];
            }
        } else if (k == 272) {
            s = (m == 0) ? U1_0[i1] * w1a : U1_1[(m-1)*16 + i1] * w1b;
        }
        T[idx] = (__bf16)s;
    }
}

// =====================================================================
// Kernel 2: group node indices by species (argmax of one-hot), padded to
// multiples of 16 with -1 sentinels. Deterministic serial scan per species.
// =====================================================================
__global__ __launch_bounds__(32)
void build_species(const float* __restrict__ attrs, int* __restrict__ list,
                   int* __restrict__ cnt)
{
    int e = threadIdx.x;
    if (e >= En) return;
    int n = 0;
    for (int b = 0; b < Bn; ++b)
        if (attrs[b*En + e] > 0.5f) list[e*Bn + n++] = b;
    while (n & 15) list[e*Bn + n++] = -1;
    cnt[e] = n;                        // padded count
}

// =====================================================================
// Kernel 3: main contraction. 1 wave per (c, node-tile, e).
//   A[16 x 288] = per-node [x⊗x | x | 1 | 0]  (bf16, built in LDS)
//   B[288 x 64] = species table               (bf16, streamed from L2)
//   D = A·B  via 4 N-tiles x 9 K-steps of v_wmma_f32_16x16x32_bf16
//   epilogue: out[b,m] = Σ_i1 D[b,(m,i1)]·x[b,i1]  (LDS transpose reduce)
// =====================================================================
__global__ __launch_bounds__(32)
void mace_contract(const float* __restrict__ a_i, const __bf16* __restrict__ tbl,
                   const int* __restrict__ list, const int* __restrict__ cnt,
                   float* __restrict__ out)
{
    const int c = blockIdx.x;
    const int t = blockIdx.y;
    const int e = blockIdx.z;
    if (t * 16 >= cnt[e]) return;      // uniform early exit

    __shared__ float  Xf[16][16];
    __shared__ __attribute__((aligned(16))) __bf16 XX[16][KT];   // 9216 B
    __shared__ int    rows[16];

    // after the A-fragments are in registers, XX is dead: reuse as f32 scratch
    float* Ep = (float*)&XX[0][0];     // 16 rows x 16 i1

    const int lane = threadIdx.x;

    if (lane < 16) rows[lane] = list[e*Bn + t*16 + lane];
    __syncthreads();

    // ---- load X tile: 2 lanes per row, 8 floats each ----
    {
        int r = lane >> 1, h = lane & 1;
        int bl = rows[r];
        float4 v0 = make_float4(0.f,0.f,0.f,0.f), v1 = v0;
        if (bl >= 0) {
            const float4* p = (const float4*)(a_i + ((size_t)bl*Cc + c)*DIMI + h*8);
            v0 = p[0]; v1 = p[1];
        }
        *(float4*)&Xf[r][h*8]     = v0;
        *(float4*)&Xf[r][h*8 + 4] = v1;
    }
    __syncthreads();

    // ---- build augmented K-vector [x⊗x | x | 1 | 0] in bf16 ----
    for (int idx = lane; idx < 16 * KT; idx += 32) {
        int r = idx / KT, k = idx % KT;
        float v;
        if      (k < 256) v = Xf[r][k >> 4] * Xf[r][k & 15];
        else if (k < 272) v = Xf[r][k - 256];
        else if (k == 272) v = 1.0f;
        else               v = 0.0f;
        XX[r][k] = (__bf16)v;
    }
    __syncthreads();

    // ---- A fragments (ISA 16x32 bf16 layout) ----
    v16bf afrag[KS];
    {
        int M  = lane & 15;
        int hi = (lane >> 4) * 8;      // lanes 16-31 hold K+8 halves
        #pragma unroll
        for (int j = 0; j < KS; ++j) {
            union { v16bf v; int4 q[2]; } u;
            u.q[0] = *(const int4*)&XX[M][32*j + hi];
            u.q[1] = *(const int4*)&XX[M][32*j + 16 + hi];
            afrag[j] = u.v;
        }
    }

    const int coln  = lane & 15;                // B col = N = i1 in epilogue
    const int rbase = (lane >> 4) * 8;          // D rows r+rbase

    // ---- hoist x[row, i1=coln] for this lane's 8 D-rows (m-invariant) ----
    float xr[8];
    #pragma unroll
    for (int r = 0; r < 8; ++r) xr[r] = Xf[r + rbase][coln];

    __syncthreads();                   // XX reads done; safe to recycle as Ep

    // epilogue reader assignment: 2 lanes per row
    const int erow  = lane >> 1;
    const int ehalf = lane & 1;

    // ---- 4 N-tiles (m = l0, l1_m0..2), 9 K-steps each ----
    const __bf16* tbase = tbl + (size_t)(e*Cc + c) * (64 * KT);
    const int koff = (lane >> 4) * 8;           // lanes 16-31 hold K+8 halves

    #pragma unroll
    for (int m = 0; m < 4; ++m) {
        v8f acc = {0.f,0.f,0.f,0.f,0.f,0.f,0.f,0.f};
        const __bf16* tb = tbase + (size_t)(m*16 + coln) * KT + koff;
        #pragma unroll
        for (int j = 0; j < KS; ++j) {
            union { v16bf v; int4 q[2]; } u;
            const int4* p = (const int4*)(tb + 32*j);
            u.q[0] = p[0]; u.q[1] = p[2];
            acc = __builtin_amdgcn_wmma_f32_16x16x32_bf16(
                      false, afrag[j], false, u.v, (short)0, acc, false, false);
        }
        // ---- epilogue: out[row,m] = Σ_i1 D[row,(m,i1)] * x[row,i1] ----
        // D layout: VGPR r holds row (r + 8*(lane>=16)), N = lane%16 = i1.
        // Transpose through LDS, then row-sum with contiguous b128 reads.
        #pragma unroll
        for (int r = 0; r < 8; ++r)
            Ep[(r + rbase)*16 + coln] = acc[r] * xr[r];
        __syncthreads();
        {
            const float4* p = (const float4*)&Ep[erow*16 + ehalf*8];
            float4 a = p[0], b = p[1];
            float s = ((a.x + a.y) + (a.z + a.w)) + ((b.x + b.y) + (b.z + b.w));
            // combine the two half-row sums: ds_swizzle SWAPX1 (xor 1), no clamps
            s += __int_as_float(
                     __builtin_amdgcn_ds_swizzle(__float_as_int(s), 0x041F));
            if (ehalf == 0) {
                int bl = rows[erow];
                if (bl >= 0) {
                    int oi = (m == 0) ? bl*512 + c
                                      : bl*512 + 128 + c*3 + (m - 1);
                    out[oi] = s;
                }
            }
        }
        __syncthreads();               // WAR: next m reuses Ep
    }
}

// =====================================================================
extern "C" void kernel_launch(void* const* d_in, const int* in_sizes, int n_in,
                              void* d_out, int out_size, void* d_ws, size_t ws_size,
                              hipStream_t stream)
{
    (void)in_sizes; (void)n_in; (void)out_size; (void)ws_size;

    const float* a_i   = (const float*)d_in[0];
    const float* attrs = (const float*)d_in[1];
    const float* U3_0  = (const float*)d_in[2];
    const float* U2_0  = (const float*)d_in[3];
    const float* U1_0  = (const float*)d_in[4];
    const float* W3_0  = (const float*)d_in[5];
    const float* W2_0  = (const float*)d_in[6];
    const float* W1_0  = (const float*)d_in[7];
    const float* U3_1  = (const float*)d_in[8];
    const float* U2_1  = (const float*)d_in[9];
    const float* U1_1  = (const float*)d_in[10];
    const float* W3_1  = (const float*)d_in[11];
    const float* W2_1  = (const float*)d_in[12];
    const float* W1_1  = (const float*)d_in[13];

    // workspace layout: [bf16 table | species lists | counts]
    size_t tbl_elems = (size_t)En * Cc * 64 * KT;          // 23,592,960
    __bf16* tbl = (__bf16*)d_ws;
    int*    list = (int*)((char*)d_ws + tbl_elems * sizeof(__bf16));
    int*    cnt  = list + (size_t)En * Bn;

    hipLaunchKernelGGL(build_table, dim3(En * Cc), dim3(256), 0, stream,
                       U3_0, U2_0, U1_0, W3_0, W2_0, W1_0,
                       U3_1, U2_1, U1_1, W3_1, W2_1, W1_1, tbl);

    hipLaunchKernelGGL(build_species, dim3(1), dim3(32), 0, stream,
                       attrs, list, cnt);

    // worst case: all nodes one species -> 64 tiles; others early-exit.
    hipLaunchKernelGGL(mace_contract, dim3(Cc, Bn / 16, En), dim3(32), 0, stream,
                       a_i, tbl, list, cnt, (float*)d_out);
}
